// AdaAttention_64639257805318
// MI455X (gfx1250) — matrix operand order repaired and checked by reading, not compiled, not run on hardware
//
#include <hip/hip_runtime.h>

// ---------------------------------------------------------------------------
// AdaAttention for MI455X (gfx1250): bf16 WMMA GEMMs + flash attention.
// B=2, C=512, T=128, J=24 -> N=3072 tokens, H=8 heads, d=64.
// Flash uses the S^T formulation: softmax columns are lane-local, so row
// stats need only one xor-16 shuffle each and P^T -> B-operand needs only
// 4 shuffles (no LDS round trip).  Straight loop: intra-wave WMMA hazard
// slots are hidden by the many resident waves, so we minimize dynamic
// instructions per wave instead of hand-pipelining.
// ---------------------------------------------------------------------------

namespace {
constexpr int BB = 2;
constexpr int C  = 512;
constexpr int T_ = 128;
constexpr int J_ = 24;
constexpr int H_ = 8;
constexpr int N_ = T_ * J_;   // 3072 tokens
constexpr float EPS = 1e-5f;
}

typedef __attribute__((ext_vector_type(16))) __bf16        v16bf;
typedef __attribute__((ext_vector_type(8)))  float         v8f;
typedef __attribute__((ext_vector_type(4)))  unsigned int  u32x4;

union Frag {               // one WMMA 16-bit A/B operand: 32 bytes / lane
  u32x4 u[2];
  v16bf v;
};

static __device__ __forceinline__ unsigned short f2bf(float f) {
  unsigned int u = __float_as_uint(f);
  u += 0x7FFFu + ((u >> 16) & 1u);            // round-to-nearest-even
  return (unsigned short)(u >> 16);
}

static __device__ __forceinline__ unsigned int bfpack(float a, float b) {
  unsigned int ua = __float_as_uint(a);
  ua += 0x7FFFu + ((ua >> 16) & 1u);
  unsigned int ub = __float_as_uint(b);
  ub += 0x7FFFu + ((ub >> 16) & 1u);
  return (ua >> 16) | (ub & 0xFFFF0000u);
}

static __device__ __forceinline__ v8f vzero() {
  v8f z = {0.f, 0.f, 0.f, 0.f, 0.f, 0.f, 0.f, 0.f};
  return z;
}

static __device__ __forceinline__ v8f wmma_bf16(const Frag& a, const Frag& b, v8f c) {
  // (neg_a, A, neg_b, B, c_mod, C, reuse_a, reuse_b)
  return __builtin_amdgcn_wmma_f32_16x16x32_bf16(false, a.v, false, b.v,
                                                 (short)0, c, false, false);
}

// ---------------------------------------------------------------------------
// Instance-norm (optional) + fp32 -> bf16, channel-major in -> token-major out.
// One block per (b,c) row; biased variance over N tokens.
// ---------------------------------------------------------------------------
__global__ __launch_bounds__(256) void k_instnorm_tok_bf16(
    const float* __restrict__ in,          // [B][C][N]
    unsigned short* __restrict__ out_t,    // [B][N][C] bf16
    int doNorm)
{
  const int row  = blockIdx.x;             // b*C + c
  const int b    = row / C;
  const int c    = row % C;
  const int lane = threadIdx.x & 31;
  const int wave = threadIdx.x >> 5;
  const float* p = in + (size_t)row * N_;

  float s = 0.f, ss = 0.f;
  for (int i = threadIdx.x; i < N_; i += 256) {
    float v = p[i];
    s += v; ss += v * v;
  }
  #pragma unroll
  for (int off = 16; off >= 1; off >>= 1) {
    s  += __shfl_xor(s,  off, 32);
    ss += __shfl_xor(ss, off, 32);
  }
  __shared__ float rs[8], rss[8];
  if (lane == 0) { rs[wave] = s; rss[wave] = ss; }
  __syncthreads();
  float tot = 0.f, tot2 = 0.f;
  #pragma unroll
  for (int w = 0; w < 8; ++w) { tot += rs[w]; tot2 += rss[w]; }

  const float mean = tot / (float)N_;
  const float var  = tot2 / (float)N_ - mean * mean;
  const float rstd = doNorm ? rsqrtf(fmaxf(var, 0.f) + EPS) : 1.0f;
  const float mu   = doNorm ? mean : 0.0f;

  for (int i = threadIdx.x; i < N_; i += 256)
    out_t[((size_t)b * N_ + i) * C + c] = f2bf((p[i] - mu) * rstd);
}

// ---------------------------------------------------------------------------
// fp32 -> bf16 (weights)
// ---------------------------------------------------------------------------
__global__ __launch_bounds__(256) void k_f32_to_bf16(
    const float* __restrict__ in, unsigned short* __restrict__ out, int n)
{
  int i = blockIdx.x * 256 + threadIdx.x;
  if (i < n) out[i] = f2bf(in[i]);
}

// ---------------------------------------------------------------------------
// 1x1 conv as GEMM: out[b,o,n] = (sum_c W[o,c]*act[b,n,c] + bias[o]) * oscale
//   A = W tile (16o x 32k), row-major K contiguous
//   B = act tile (32k x 16n), token-major -> per-lane two 16B chunks
// Block 256 = 8 waves (2x4), block tile 64o x 128n, wave tile 32x32.
// MODE 0: token-major bf16 out   [B][N][C]     (q with oscale=1/8, k)
// MODE 1: channel-major bf16 out [B][C][N]     (v)
// MODE 2: channel-major fp32 out + residual    (final projection)
// ---------------------------------------------------------------------------
template <int MODE>
__global__ __launch_bounds__(256) void k_gemm1x1(
    const unsigned short* __restrict__ Wbf,   // [512][512] bf16
    const float*          __restrict__ bias,  // [512]
    const unsigned short* __restrict__ act,   // [B][N][512] bf16 token-major
    void*                 __restrict__ outp,
    const float*          __restrict__ resid, // [B][512][N] fp32 (MODE 2)
    float                 oscale)
{
  constexpr int M = C, K = C, NC = N_;
  const int b    = blockIdx.z;
  const int lane = threadIdx.x & 31;
  const int wave = threadIdx.x >> 5;
  const int lc   = lane & 15;
  const int hi   = lane >> 4;
  const int o_base = blockIdx.x * 64  + (wave >> 2) * 32;
  const int n_base = blockIdx.y * 128 + (wave & 3) * 32;
  const unsigned short* actb = act + (size_t)b * NC * K;

  v8f acc[2][2];
  #pragma unroll
  for (int r = 0; r < 2; ++r)
    #pragma unroll
    for (int s = 0; s < 2; ++s) acc[r][s] = vzero();

  for (int k0 = 0; k0 < K; k0 += 32) {
    Frag af[2], bfr[2];
    #pragma unroll
    for (int r = 0; r < 2; ++r) {                 // A: two 16B chunks
      const unsigned short* ap = Wbf + (size_t)(o_base + r * 16 + lc) * K + k0 + hi * 8;
      af[r].u[0] = *(const u32x4*)ap;
      af[r].u[1] = *(const u32x4*)(ap + 16);
      __builtin_prefetch((const void*)(ap + 32), 0, 1);   // global_prefetch_b8
    }
    #pragma unroll
    for (int s = 0; s < 2; ++s) {                 // B: two 16B chunks
      const unsigned short* bp = actb + (size_t)(n_base + s * 16 + lc) * K + k0 + hi * 8;
      bfr[s].u[0] = *(const u32x4*)bp;
      bfr[s].u[1] = *(const u32x4*)(bp + 16);
      __builtin_prefetch((const void*)(bp + 32), 0, 1);
    }
    #pragma unroll
    for (int r = 0; r < 2; ++r)
      #pragma unroll
      for (int s = 0; s < 2; ++s)
        acc[r][s] = wmma_bf16(af[r], bfr[s], acc[r][s]);
  }

  // Epilogue.  C/D layout: col n = lc, row o = v + hi*8.
  #pragma unroll
  for (int r = 0; r < 2; ++r) {
    #pragma unroll
    for (int s = 0; s < 2; ++s) {
      const int n = n_base + s * 16 + lc;
      #pragma unroll
      for (int v = 0; v < 8; ++v) {
        const int o = o_base + r * 16 + v + hi * 8;
        float val = acc[r][s][v] + bias[o];
        if (MODE == 0) {
          ((unsigned short*)outp)[((size_t)b * NC + n) * M + o] = f2bf(val * oscale);
        } else if (MODE == 1) {
          ((unsigned short*)outp)[((size_t)b * M + o) * NC + n] = f2bf(val * oscale);
        } else {
          const size_t idx = ((size_t)b * M + o) * NC + n;
          ((float*)outp)[idx] = val + resid[idx];
        }
      }
    }
  }
}

// ---------------------------------------------------------------------------
// Flash attention, one wave per (b, h, 16-query block).  S^T formulation:
//   S^T (32m x 16n) = K-tiles(A, rows m) x Q(B, cols n)  [chained K=32 WMMAs]
//   softmax columns are lane-local (col n = lane&15, halves duplicated):
//   in-lane max/sum over 16 values + one xor-16 shuffle each.
//   O (64d x 16n)  += V(A, rows d) x P^T(B, cols n); P^T C/D->B operand via
//   bf16 packing + 4 xor-16 shuffles.  No LDS, no per-row broadcasts.
// ---------------------------------------------------------------------------
__global__ __launch_bounds__(128) void k_flash_attn(
    const unsigned short* __restrict__ q_t,   // [B][N][C] bf16 (pre-scaled 1/8)
    const unsigned short* __restrict__ k_t,   // [B][N][C] bf16
    const unsigned short* __restrict__ v_cm,  // [B][C][N] bf16
    unsigned short*       __restrict__ o_t)   // [B][N][C] bf16
{
  const int lane = threadIdx.x & 31;
  const int widx = threadIdx.x >> 5;
  const int lc   = lane & 15;
  const int hi   = lane >> 4;

  const int w     = blockIdx.x * 4 + widx;
  const int nblks = N_ / 16;                  // 192
  const int bh    = w / nblks;
  const int nblk  = w % nblks;
  const int b     = bh >> 3;
  const int h     = bh & 7;
  const int n0    = nblk * 16;

  const unsigned short* qb = q_t  + ((size_t)b * N_) * C + h * 64;
  const unsigned short* kb = k_t  + ((size_t)b * N_) * C + h * 64;
  const unsigned short* vb = v_cm + ((size_t)b * C + h * 64) * N_;

  // Q as B operand (col n = n0+lc, K = d): two 16B chunks per frag
  Frag qf[2];
  #pragma unroll
  for (int j = 0; j < 2; ++j) {
    const unsigned short* p = qb + (size_t)(n0 + lc) * C + j * 32 + hi * 8;
    qf[j].u[0] = *(const u32x4*)p;
    qf[j].u[1] = *(const u32x4*)(p + 16);
  }

  v8f accO[4];
  #pragma unroll
  for (int j = 0; j < 4; ++j) accO[j] = vzero();
  float mi = -3.0e38f, li = 0.f;              // lane-local: column n = n0+lc

  for (int m0 = 0; m0 < N_; m0 += 32) {
    // ---- S^T tiles: tile s covers key rows m0+s*16 .. +15
    v8f St[2];
    #pragma unroll
    for (int s = 0; s < 2; ++s) {
      const unsigned short* kp = kb + (size_t)(m0 + s * 16 + lc) * C + hi * 8;
      Frag ka0, ka1;                          // A layout: two 16B chunks
      ka0.u[0] = *(const u32x4*)kp;
      ka0.u[1] = *(const u32x4*)(kp + 16);
      ka1.u[0] = *(const u32x4*)(kp + 32);
      ka1.u[1] = *(const u32x4*)(kp + 48);
      v8f t = wmma_bf16(ka0, qf[0], vzero());
      St[s]  = wmma_bf16(ka1, qf[1], t);
    }

    // ---- lane-local softmax over the 32 key rows of column n0+lc
    float mx = St[0][0];
    #pragma unroll
    for (int v = 1; v < 8; ++v) mx = fmaxf(mx, St[0][v]);
    #pragma unroll
    for (int v = 0; v < 8; ++v) mx = fmaxf(mx, St[1][v]);
    mx = fmaxf(mx, __shfl_xor(mx, 16, 32));   // combine the two 16-lane halves
    const float mnew = fmaxf(mi, mx);

    float rsum = 0.f;
    #pragma unroll
    for (int s = 0; s < 2; ++s)
      #pragma unroll
      for (int v = 0; v < 8; ++v) {
        float p = __expf(St[s][v] - mnew);
        St[s][v] = p;
        rsum += p;
      }
    rsum += __shfl_xor(rsum, 16, 32);

    const float corr = __expf(mi - mnew);
    li = li * corr + rsum;
    mi = mnew;
    #pragma unroll
    for (int j = 0; j < 4; ++j)
      #pragma unroll
      for (int v = 0; v < 8; ++v) accO[j][v] *= corr;   // lane-local rescale

    // ---- P^T -> bf16 B operand (cols n).  Each lane keeps its half's 8
    // values and swaps the other 8 with its xor-16 partner (4 dwords).
    Frag pb;
    #pragma unroll
    for (int i = 0; i < 4; ++i) {
      unsigned int pa = bfpack(St[0][2 * i], St[0][2 * i + 1]);  // rows m0+2i(+1)  (+8*hi)
      unsigned int pc = bfpack(St[1][2 * i], St[1][2 * i + 1]);  // rows m0+16+2i(+1)
      unsigned int send = hi ? pa : pc;       // what the partner needs
      unsigned int recv = __shfl_xor(send, 16, 32);
      pb.u[0][i] = hi ? recv : pa;            // K = 0..7  half
      pb.u[1][i] = hi ? pc   : recv;          // K = 8..15 half
    }

    // ---- O(64d x 16n) += V(A, rows d) x P^T(B)   (4 d-tiles, K = 32 keys)
    #pragma unroll
    for (int j = 0; j < 4; ++j) {
      const unsigned short* vp = vb + (size_t)(j * 16 + lc) * N_ + m0 + hi * 16;
      Frag va;
      va.u[0] = *(const u32x4*)vp;
      va.u[1] = *(const u32x4*)(vp + 8);
      accO[j] = wmma_bf16(va, pb, accO[j]);
    }
  }

  // ---- normalize (lane-local 1/l) and store O token-major bf16
  const float inv = 1.0f / li;
  const int   n   = n0 + lc;
  #pragma unroll
  for (int j = 0; j < 4; ++j)
    #pragma unroll
    for (int v = 0; v < 8; ++v) {
      const int d = j * 16 + v + hi * 8;      // C/D row
      o_t[((size_t)b * N_ + n) * C + h * 64 + d] = f2bf(accO[j][v] * inv);
    }
}

// ---------------------------------------------------------------------------
// Host launcher
// ---------------------------------------------------------------------------
extern "C" void kernel_launch(void* const* d_in, const int* in_sizes, int n_in,
                              void* d_out, int out_size, void* d_ws, size_t ws_size,
                              hipStream_t stream) {
  const float* x    = (const float*)d_in[0];
  const float* s    = (const float*)d_in[1];
  const float* Wq_w = (const float*)d_in[2];
  const float* Wq_b = (const float*)d_in[3];
  const float* Wk_w = (const float*)d_in[4];
  const float* Wk_b = (const float*)d_in[5];
  const float* Wv_w = (const float*)d_in[6];
  const float* Wv_b = (const float*)d_in[7];
  const float* Wo_w = (const float*)d_in[8];
  const float* Wo_b = (const float*)d_in[9];

  const size_t act_bytes = (size_t)BB * N_ * C * sizeof(unsigned short); // 6.29 MB
  const size_t w_bytes   = (size_t)C * C * sizeof(unsigned short);       // 0.5 MB
  char* ws = (char*)d_ws;
  unsigned short* xn_t  = (unsigned short*)(ws + 0 * act_bytes);
  unsigned short* sn_t  = (unsigned short*)(ws + 1 * act_bytes);
  unsigned short* sr_t  = (unsigned short*)(ws + 2 * act_bytes);
  unsigned short* q_t   = (unsigned short*)(ws + 3 * act_bytes);
  unsigned short* k_t   = (unsigned short*)(ws + 4 * act_bytes);
  unsigned short* v_cm  = (unsigned short*)(ws + 5 * act_bytes);
  unsigned short* o_t   = (unsigned short*)(ws + 6 * act_bytes);
  unsigned short* Wq_bf = (unsigned short*)(ws + 7 * act_bytes + 0 * w_bytes);
  unsigned short* Wk_bf = (unsigned short*)(ws + 7 * act_bytes + 1 * w_bytes);
  unsigned short* Wv_bf = (unsigned short*)(ws + 7 * act_bytes + 2 * w_bytes);
  unsigned short* Wo_bf = (unsigned short*)(ws + 7 * act_bytes + 3 * w_bytes);

  // 1) activations: instance-norm (x, s) + raw s, to token-major bf16
  k_instnorm_tok_bf16<<<BB * C, 256, 0, stream>>>(x, xn_t, 1);
  k_instnorm_tok_bf16<<<BB * C, 256, 0, stream>>>(s, sn_t, 1);
  k_instnorm_tok_bf16<<<BB * C, 256, 0, stream>>>(s, sr_t, 0);

  // 2) weights -> bf16
  const int wn = C * C;
  k_f32_to_bf16<<<(wn + 255) / 256, 256, 0, stream>>>(Wq_w, Wq_bf, wn);
  k_f32_to_bf16<<<(wn + 255) / 256, 256, 0, stream>>>(Wk_w, Wk_bf, wn);
  k_f32_to_bf16<<<(wn + 255) / 256, 256, 0, stream>>>(Wv_w, Wv_bf, wn);
  k_f32_to_bf16<<<(wn + 255) / 256, 256, 0, stream>>>(Wo_w, Wo_bf, wn);

  // 3) q/k/v projections (WMMA GEMMs); 1/sqrt(d) folded into q
  dim3 ggrid(C / 64, N_ / 128, BB);
  k_gemm1x1<0><<<ggrid, 256, 0, stream>>>(Wq_bf, Wq_b, xn_t, (void*)q_t,  nullptr, 0.125f);
  k_gemm1x1<0><<<ggrid, 256, 0, stream>>>(Wk_bf, Wk_b, sn_t, (void*)k_t,  nullptr, 1.0f);
  k_gemm1x1<1><<<ggrid, 256, 0, stream>>>(Wv_bf, Wv_b, sr_t, (void*)v_cm, nullptr, 1.0f);

  // 4) flash attention: 2*8*192 = 3072 waves, 4 waves per block
  k_flash_attn<<<(BB * H_ * (N_ / 16)) / 4, 128, 0, stream>>>(q_t, k_t, v_cm, o_t);

  // 5) output projection + bias + residual, fp32
  k_gemm1x1<2><<<ggrid, 256, 0, stream>>>(Wo_bf, Wo_b, o_t, d_out, x, 1.0f);
}
